// SpikeSelfAttention_37692632989891
// MI455X (gfx1250) — 8-bit matrix operand order repaired and checked by reading, compiled, not run on hardware
//
#include <hip/hip_runtime.h>
#include <hip/hip_fp16.h>


typedef __attribute__((ext_vector_type(16))) _Float16 v16h;
typedef __attribute__((ext_vector_type(8)))  _Float16 v8h;
typedef __attribute__((ext_vector_type(4)))  _Float16 v4h;
typedef __attribute__((ext_vector_type(4)))  float    v4f;
typedef __attribute__((ext_vector_type(8)))  float    v8f;
typedef __attribute__((ext_vector_type(8)))  int      v8i;

#define T_ 4
#define B_ 16
#define N_ 256
#define C_ 512
#define H_ 8
#define D_ 64
#define BN_EPS 1e-5f

__device__ __forceinline__ void async_b128_to_lds(unsigned lds_off, const void* gaddr) {
  asm volatile("global_load_async_to_lds_b128 %0, %1, off" ::"v"(lds_off), "v"(gaddr)
               : "memory");
}
__device__ __forceinline__ void wait_asynccnt0() {
  asm volatile("s_wait_asynccnt 0x0" ::: "memory");
}

__device__ __forceinline__ v16h load_frag_h16(const _Float16* __restrict__ row, int kb) {
  v8h lo = *(const v8h*)(row + kb);
  v8h hi = *(const v8h*)(row + 16 + kb);
  return __builtin_shufflevector(lo, hi, 0, 1, 2, 3, 4, 5, 6, 7, 8, 9, 10, 11, 12, 13, 14, 15);
}
__device__ __forceinline__ int a8_koff(int j, int lanehi8  ) {
  return (j & 1) * 4 + ((j >> 1) & 1) * 16 + (j >> 2) * 32 + lanehi8;
}
__device__ __forceinline__ int b8_koff(int j, int lanehi16  ) {
  return (j & 1) * 4 + ((j >> 1) & 1) * 16 + (j >> 2) * 32 + lanehi16 / 2;
}

__global__ __launch_bounds__(256) void k_cvt_f16(const float* __restrict__ s,
                                                 _Float16* __restrict__ d, int n) {
  int i = (blockIdx.x * 256 + threadIdx.x) * 4;
  if (i >= n) return;
  v4f v = *(const v4f*)(s + i);
  v4h h;
#pragma unroll
  for (int e = 0; e < 4; ++e) h[e] = (_Float16)v[e];
  *(v4h*)(d + i) = h;
}

__global__ __launch_bounds__(128) void k_linear_bn_lif(
    const _Float16* __restrict__ xh, const _Float16* __restrict__ wh,
    const float* __restrict__ bn_scale, const float* __restrict__ bn_bias,
    const float* __restrict__ bn_mean, const float* __restrict__ bn_var,
    int branch, signed char* __restrict__ S, signed char* __restrict__ ST) {
  __shared__ _Float16 smA[2][16 * 32];

  const int lane  = threadIdx.x & 31;
  const int wave  = threadIdx.x >> 5;
  const int n0    = blockIdx.x * 16;
  const int b     = blockIdx.y;
  const int c0    = blockIdx.z * 256 + wave * 64;
  const int nlo   = lane & 15;
  const int kb    = ((lane >> 4) & 1) * 8;
  const int rowhi = (lane >> 4) * 8;
  const int sr  = lane >> 2;
  const int sc  = (lane & 3) * 8;
  const unsigned lds0 = (unsigned)(uintptr_t)&smA[0][0] + (unsigned)lane * 16;

  float premul[4], mu[4], beta[4];
#pragma unroll
  for (int ti = 0; ti < 4; ++ti) {
    int c = c0 + 16 * ti + nlo;
    premul[ti] = bn_scale[branch * C_ + c] * rsqrtf(bn_var[branch * C_ + c] + BN_EPS);
    mu[ti]     = bn_mean[branch * C_ + c];
    beta[ti]   = bn_bias[branch * C_ + c];
  }

  float vmem[4][8];
#pragma unroll
  for (int ti = 0; ti < 4; ++ti)
#pragma unroll
    for (int r = 0; r < 8; ++r) vmem[ti][r] = 0.0f;

  for (int t = 0; t < T_; ++t) {
    const _Float16* xt = xh + (size_t)((t * B_ + b) * N_ + n0) * C_;
    const _Float16* g1 = xt + (size_t)sr * C_ + sc;
    const _Float16* g2 = xt + (size_t)(sr + 8) * C_ + sc;
    if (wave == 0) {
      async_b128_to_lds(lds0, g1);
      async_b128_to_lds(lds0 + 512, g2);
    }
    v8f acc[4] = {v8f{}, v8f{}, v8f{}, v8f{}};
    for (int ck = 0; ck < C_ / 32; ++ck) {
      int sel = ck & 1;
      if (wave == 0) wait_asynccnt0();
      __syncthreads();
      if (wave == 0 && ck + 1 < C_ / 32) {
        async_b128_to_lds(lds0 + (sel ^ 1) * 1024, g1 + (ck + 1) * 32);
        async_b128_to_lds(lds0 + (sel ^ 1) * 1024 + 512, g2 + (ck + 1) * 32);
      }
      v16h a = load_frag_h16(&smA[sel][0] + nlo * 32, kb);
#pragma unroll
      for (int ti = 0; ti < 4; ++ti) {
        const _Float16* wrow = wh + (size_t)(c0 + 16 * ti + nlo) * C_ + ck * 32;
        if (ck + 2 < C_ / 32) __builtin_prefetch(wrow + 64, 0, 0);
        v16h bf = load_frag_h16(wrow, kb);
        acc[ti] = __builtin_amdgcn_wmma_f32_16x16x32_f16(
            false, a, false, bf, (short)0, acc[ti], false, false);
      }
      __syncthreads();
    }
#pragma unroll
    for (int ti = 0; ti < 4; ++ti) {
      int c = c0 + 16 * ti + nlo;
#pragma unroll
      for (int r = 0; r < 8; ++r) {
        float yn = (acc[ti][r] - mu[ti]) * premul[ti] + beta[ti];
        float v  = vmem[ti][r] + (yn - vmem[ti][r]) * 0.5f;
        float s  = (v >= 1.0f) ? 1.0f : 0.0f;
        int   n  = n0 + r + rowhi;
        if (S)  S[(size_t)((t * B_ + b) * N_ + n) * C_ + c] = (signed char)s;
        if (ST) ST[(size_t)((t * B_ + b) * C_ + c) * N_ + n] = (signed char)s;
        vmem[ti][r] = v * (1.0f - s);
      }
    }
  }
}

__global__ __launch_bounds__(128) void k_qk_lif(
    const signed char* __restrict__ Sq, const signed char* __restrict__ Sk,
    signed char* __restrict__ Sattn) {
  const int lane  = threadIdx.x & 31;
  const int wave  = threadIdx.x >> 5;
  const int n0    = blockIdx.x * 16;
  const int h     = blockIdx.y;
  const int b     = blockIdx.z;
  const int nlo   = lane & 15;
  const int la8   = (lane >= 16) ? 8 : 0;
  const int lb16  = (lane >= 16) ? 16 : 0;
  const int rowhi = (lane >> 4) * 8;
  const int m0w   = wave * 64;

  float vmem[4][8];
#pragma unroll
  for (int ti = 0; ti < 4; ++ti)
#pragma unroll
    for (int r = 0; r < 8; ++r) vmem[ti][r] = 0.0f;

  for (int t = 0; t < T_; ++t) {
    const signed char* qrow = Sq + (size_t)((t * B_ + b) * N_ + (n0 + nlo)) * C_ + h * D_;
    v8i a;
#pragma unroll
    for (int j = 0; j < 8; ++j) a[j] = *(const int*)(qrow + a8_koff(j, la8));
#pragma unroll
    for (int ti = 0; ti < 4; ++ti) {
      int m0 = m0w + 16 * ti;
      const signed char* krow = Sk + (size_t)((t * B_ + b) * N_ + (m0 + nlo)) * C_ + h * D_;
      v8i bf;
#pragma unroll
      for (int j = 0; j < 8; ++j) bf[j] = *(const int*)(krow + b8_koff(j, lb16));
      v8i cz = v8i{};
      v8i cnt = __builtin_amdgcn_wmma_i32_16x16x64_iu8(false, a, false, bf, cz, false, false);
#pragma unroll
      for (int r = 0; r < 8; ++r) {
        float yn = (float)cnt[r] * 0.125f;
        float v  = vmem[ti][r] + (yn - vmem[ti][r]) * 0.5f;
        float s  = (v >= 1.0f) ? 1.0f : 0.0f;
        int   n  = n0 + r + rowhi, m = m0 + nlo;
        Sattn[(((size_t)(t * B_ + b) * H_ + h) * N_ + n) * N_ + m] = (signed char)s;
        vmem[ti][r] = v * (1.0f - s);
      }
    }
  }
}

__global__ __launch_bounds__(128) void k_av(
    const signed char* __restrict__ Sattn, const signed char* __restrict__ SvT,
    _Float16* __restrict__ outh) {
  const int lane  = threadIdx.x & 31;
  const int wave  = threadIdx.x >> 5;
  const int n0    = blockIdx.x * 16;
  const int tbh   = blockIdx.y;
  const int t     = tbh / (B_ * H_);
  const int b     = (tbh / H_) % B_;
  const int h     = tbh % H_;
  const int nlo   = lane & 15;
  const int la8   = (lane >= 16) ? 8 : 0;
  const int lb16  = (lane >= 16) ? 16 : 0;
  const int rowhi = (lane >> 4) * 8;
  const int d     = wave * 16 + nlo;

  v8i acc = v8i{};
  const signed char* arowbase =
      Sattn + (((size_t)(t * B_ + b) * H_ + h) * N_ + (n0 + nlo)) * N_;
  const signed char* browbase =
      SvT + ((size_t)(t * B_ + b) * C_ + (h * D_ + d)) * N_;
  for (int mc = 0; mc < N_; mc += 64) {
    v8i a, bf;
#pragma unroll
    for (int j = 0; j < 8; ++j) a[j] = *(const int*)(arowbase + mc + a8_koff(j, la8));
#pragma unroll
    for (int j = 0; j < 8; ++j) bf[j] = *(const int*)(browbase + mc + b8_koff(j, lb16));
    acc = __builtin_amdgcn_wmma_i32_16x16x64_iu8(false, a, false, bf, acc, false, false);
  }
#pragma unroll
  for (int r = 0; r < 8; ++r) {
    int n = n0 + r + rowhi;
    outh[(size_t)((t * B_ + b) * N_ + n) * C_ + h * D_ + d] = (_Float16)(float)acc[r];
  }
}

__global__ __launch_bounds__(128) void k_proj_bn(
    const _Float16* __restrict__ outh, const _Float16* __restrict__ who,
    const float* __restrict__ bn_scale, const float* __restrict__ bn_bias,
    const float* __restrict__ bn_mean, const float* __restrict__ bn_var,
    float* __restrict__ out) {
  __shared__ _Float16 smA[2][16 * 32];

  const int lane  = threadIdx.x & 31;
  const int wave  = threadIdx.x >> 5;
  const int row0  = blockIdx.x * 16;
  const int c0    = blockIdx.y * 256 + wave * 64;
  const int nlo   = lane & 15;
  const int kb    = ((lane >> 4) & 1) * 8;
  const int rowhi = (lane >> 4) * 8;
  const int sr    = lane >> 2;
  const int sc    = (lane & 3) * 8;
  const unsigned lds0 = (unsigned)(uintptr_t)&smA[0][0] + (unsigned)lane * 16;

  float premul[4], mu[4], beta[4];
#pragma unroll
  for (int ti = 0; ti < 4; ++ti) {
    int c = c0 + 16 * ti + nlo;
    premul[ti] = bn_scale[3 * C_ + c] * rsqrtf(bn_var[3 * C_ + c] + BN_EPS);
    mu[ti]     = bn_mean[3 * C_ + c];
    beta[ti]   = bn_bias[3 * C_ + c];
  }

  const _Float16* at = outh + (size_t)row0 * C_;
  const _Float16* g1 = at + (size_t)sr * C_ + sc;
  const _Float16* g2 = at + (size_t)(sr + 8) * C_ + sc;
  if (wave == 0) {
    async_b128_to_lds(lds0, g1);
    async_b128_to_lds(lds0 + 512, g2);
  }
  v8f acc[4] = {v8f{}, v8f{}, v8f{}, v8f{}};
  for (int ck = 0; ck < C_ / 32; ++ck) {
    int sel = ck & 1;
    if (wave == 0) wait_asynccnt0();
    __syncthreads();
    if (wave == 0 && ck + 1 < C_ / 32) {
      async_b128_to_lds(lds0 + (sel ^ 1) * 1024, g1 + (ck + 1) * 32);
      async_b128_to_lds(lds0 + (sel ^ 1) * 1024 + 512, g2 + (ck + 1) * 32);
    }
    v16h a = load_frag_h16(&smA[sel][0] + nlo * 32, kb);
#pragma unroll
    for (int ti = 0; ti < 4; ++ti) {
      const _Float16* wrow = who + (size_t)(c0 + 16 * ti + nlo) * C_ + ck * 32;
      if (ck + 2 < C_ / 32) __builtin_prefetch(wrow + 64, 0, 0);
      v16h bf = load_frag_h16(wrow, kb);
      acc[ti] = __builtin_amdgcn_wmma_f32_16x16x32_f16(
          false, a, false, bf, (short)0, acc[ti], false, false);
    }
    __syncthreads();
  }
#pragma unroll
  for (int ti = 0; ti < 4; ++ti) {
    int c = c0 + 16 * ti + nlo;
#pragma unroll
    for (int r = 0; r < 8; ++r) {
      int row = row0 + r + rowhi;
      out[(size_t)row * C_ + c] = (acc[ti][r] - mu[ti]) * premul[ti] + beta[ti];
    }
  }
}

extern "C" void kernel_launch(void* const* d_in, const int* in_sizes, int n_in,
                              void* d_out, int out_size, void* d_ws, size_t ws_size,
                              hipStream_t stream) {
  const float* x        = (const float*)d_in[0];
  const float* wq       = (const float*)d_in[1];
  const float* wk       = (const float*)d_in[2];
  const float* wv       = (const float*)d_in[3];
  const float* wo       = (const float*)d_in[4];
  const float* bn_scale = (const float*)d_in[5];
  const float* bn_bias  = (const float*)d_in[6];
  const float* bn_mean  = (const float*)d_in[7];
  const float* bn_var   = (const float*)d_in[8];

  const size_t nX    = (size_t)T_ * B_ * N_ * C_;
  const size_t nW    = (size_t)C_ * C_;
  const size_t nAttn = (size_t)T_ * B_ * H_ * N_ * N_;

  char* p = (char*)d_ws;
  _Float16*    xh    = (_Float16*)p;    p += nX * 2;
  _Float16*    whq   = (_Float16*)p;    p += nW * 2;
  _Float16*    whk   = (_Float16*)p;    p += nW * 2;
  _Float16*    whv   = (_Float16*)p;    p += nW * 2;
  _Float16*    who   = (_Float16*)p;    p += nW * 2;
  signed char* Sq    = (signed char*)p; p += nX;
  signed char* Sk    = (signed char*)p; p += nX;
  signed char* SvT   = (signed char*)p; p += nX;
  signed char* Sattn = (signed char*)p; p += nAttn;
  _Float16*    outh  = (_Float16*)p;

  dim3 blk(128);
  k_cvt_f16<<<dim3((nX / 4 + 255) / 256), dim3(256), 0, stream>>>(x, xh, (int)nX);
  k_cvt_f16<<<dim3((nW / 4 + 255) / 256), dim3(256), 0, stream>>>(wq, whq, (int)nW);
  k_cvt_f16<<<dim3((nW / 4 + 255) / 256), dim3(256), 0, stream>>>(wk, whk, (int)nW);
  k_cvt_f16<<<dim3((nW / 4 + 255) / 256), dim3(256), 0, stream>>>(wv, whv, (int)nW);
  k_cvt_f16<<<dim3((nW / 4 + 255) / 256), dim3(256), 0, stream>>>(wo, who, (int)nW);

  k_linear_bn_lif<<<dim3(N_ / 16, B_, C_ / 256), blk, 0, stream>>>(
      xh, whq, bn_scale, bn_bias, bn_mean, bn_var, 0, Sq, nullptr);
  k_linear_bn_lif<<<dim3(N_ / 16, B_, C_ / 256), blk, 0, stream>>>(
      xh, whk, bn_scale, bn_bias, bn_mean, bn_var, 1, Sk, nullptr);
  k_linear_bn_lif<<<dim3(N_ / 16, B_, C_ / 256), blk, 0, stream>>>(
      xh, whv, bn_scale, bn_bias, bn_mean, bn_var, 2, nullptr, SvT);

  k_qk_lif<<<dim3(N_ / 16, H_, B_), blk, 0, stream>>>(Sq, Sk, Sattn);

  k_av<<<dim3(N_ / 16, T_ * B_ * H_), blk, 0, stream>>>(Sattn, SvT, outh);

  k_proj_bn<<<dim3((T_ * B_ * N_) / 16, C_ / 256), blk, 0, stream>>>(
      outh, who, bn_scale, bn_bias, bn_mean, bn_var, (float*)d_out);
}
